// SNNQUT_with_clochlea_91164975825528
// MI455X (gfx1250) — 8-bit matrix operand order repaired and checked by reading, compiled, not run on hardware
//
#include <hip/hip_runtime.h>


typedef __attribute__((ext_vector_type(8))) int v8i;
typedef unsigned long long u64;
typedef unsigned char u8;

#define WMIN   0.001f
#define WMAXV  1.0f
#define QSCALE ((1.0f - 0.001f) / 31.0f)

#define B_   64
#define T_   500
#define IN_  256
#define HID_ 512
#define OUT_ 35
#define OUTP_ 48

#define NTHREADS 1024
#define NWAVES   32

#define WS_BAR  0
#define WS_S1   256
#define WS_S2   (WS_S1 + 2*B_*HID_)
#define WS_S3   (WS_S2 + 2*B_*HID_)
#define WS_Q1   (WS_S3 + 2*B_*HID_)
#define WS_Q2   (WS_Q1 + HID_*IN_)
#define WS_Q3   (WS_Q2 + HID_*HID_)
#define WS_Q4   (WS_Q3 + HID_*HID_)
#define WS_END  (WS_Q4 + OUTP_*HID_)

#define L_S       0
#define L_CHSPK   256
#define CH_STRIDE 272
#define L_CHM     (L_CHSPK + B_*CH_STRIDE)
#define L_W0      (L_CHM + B_*IN_*4)
#define L_ASPK    256
#define A_STRIDE  528
#define L_W_S     (L_ASPK + B_*A_STRIDE)
#define LDS_BYTES (L_W_S + HID_*(HID_+16))

union Frag { v8i v; u64 q[4]; };

__device__ __forceinline__ u64 ld64(const void* p) { return *(const u64*)p; }

template <int K, int NTN>
__device__ __forceinline__ void layer_step(
    const char* smem, int a_off, int a_stride,
    int w_off, const float* S, float beta, float thresh,
    float (*mem)[8], int t,
    float* outSpk, float* outMem, int Hvalid, u8* nextBuf)
{
  const int lane = threadIdx.x & 31;
  const int wave = threadIdx.x >> 5;
  const int half = lane >> 4;
  const int lcol = lane & 15;
  constexpr int NT    = 4 * NTN;
  constexpr int SLOTS = (NT + NWAVES - 1) / NWAVES;

  const int tm   = wave & 3;
  const u8* abase = (const u8*)(smem + a_off) + (tm * 16 + lcol) * a_stride;

  v8i acc[SLOTS];
#pragma unroll
  for (int s = 0; s < SLOTS; ++s) acc[s] = v8i{0, 0, 0, 0, 0, 0, 0, 0};

#pragma unroll
  for (int kc = 0; kc < K / 64; ++kc) {
    Frag a;
    const u8* ap = abase + kc * 64;
    a.q[0] = ld64(ap +  0 + half * 8);
    a.q[1] = ld64(ap + 16 + half * 8);
    a.q[2] = ld64(ap + 32 + half * 8);
    a.q[3] = ld64(ap + 48 + half * 8);
#pragma unroll
    for (int slot = 0; slot < SLOTS; ++slot) {
      const int tile = wave + slot * NWAVES;
      if (tile >= NT) break;
      const int tn = tile >> 2;
      const u8* bp = (const u8*)(smem + w_off)
                   + (tn * 16 + lcol) * (K + 16) + kc * 64;
      Frag b;
      b.q[0] = ld64(bp +  0 + half * 8);
      b.q[1] = ld64(bp + 16 + half * 8);
      b.q[2] = ld64(bp + 32 + half * 8);
      b.q[3] = ld64(bp + 48 + half * 8);
      acc[slot] = __builtin_amdgcn_wmma_i32_16x16x64_iu8(false, a.v, false, b.v,
                                                         acc[slot], false, false);
    }
  }

  float Sv[8];
#pragma unroll
  for (int r = 0; r < 8; ++r) Sv[r] = S[tm * 16 + r + 8 * half];

#pragma unroll
  for (int slot = 0; slot < SLOTS; ++slot) {
    const int tile = wave + slot * NWAVES;
    if (tile >= NT) break;
    const int tn = tile >> 2;
    const int h  = tn * 16 + lcol;
#pragma unroll
    for (int r = 0; r < 8; ++r) {
      const int m = tm * 16 + r + 8 * half;
      const float cur = WMIN * Sv[r] + QSCALE * (float)acc[slot][r];
      float mv = mem[slot][r];
      const float reset = (mv > thresh) ? thresh : 0.0f;
      mv = beta * mv + cur - reset;
      mem[slot][r] = mv;
      const float spk = (mv > thresh) ? 1.0f : 0.0f;
      if (h < Hvalid) {
        const size_t o = (size_t)(t * B_ + m) * Hvalid + h;
        outSpk[o] = spk;
        if (outMem) outMem[o] = mv;
      }
      if (nextBuf) nextBuf[m * HID_ + h] = (u8)(mv > thresh ? 1 : 0);
    }
  }
}

__global__ void quant_kernel(const float* W1, const float* W2, const float* W3,
                             const float* W4, u8* ws)
{
  const int idx = blockIdx.x * blockDim.x + threadIdx.x;
  if (idx == 0) *(unsigned*)(ws + WS_BAR) = 0u;
  const int n1 = HID_ * IN_, n2 = HID_ * HID_, n3 = HID_ * HID_, n4 = OUTP_ * HID_;
  const int total = n1 + n2 + n3 + n4;
  for (int i = idx; i < total; i += gridDim.x * blockDim.x) {
    float w; u8* dst;
    if (i < n1)                { w = W1[i];            dst = ws + WS_Q1 + i; }
    else if (i < n1 + n2)      { w = W2[i - n1];       dst = ws + WS_Q2 + (i - n1); }
    else if (i < n1 + n2 + n3) { w = W3[i - n1 - n2];  dst = ws + WS_Q3 + (i - n1 - n2); }
    else {
      const int j = i - n1 - n2 - n3;
      const int h = j / HID_;
      w = (h < OUT_) ? W4[h * HID_ + (j % HID_)] : WMIN;
      dst = ws + WS_Q4 + j;
    }
    const float wc = fminf(fmaxf(w, WMIN), WMAXV);
    *dst = (u8)(int)rintf((wc - WMIN) / QSCALE);
  }
}

__global__ void __launch_bounds__(NTHREADS, 1) snn_pipeline(
    const float* __restrict__ x, const float* __restrict__ Wch,
    const float* __restrict__ cbetas,
    const float* pb1, const float* pb2, const float* pb3, const float* pb4,
    u8* ws, float* out)
{
  extern __shared__ char smem[];
  const int stage = blockIdx.x;
  const int tid   = threadIdx.x;
  unsigned* bar   = (unsigned*)(ws + WS_BAR);
  float* S        = (float*)(smem + L_S);

  const float b1 = fminf(fmaxf(pb1[0], 0.f), 1.f);
  const float b2 = fminf(fmaxf(pb2[0], 0.f), 1.f);
  const float b3 = fminf(fmaxf(pb3[0], 0.f), 1.f);
  const float b4 = fminf(fmaxf(pb4[0], 0.f), 1.f);

  {
    int rows, K, woff; const u8* src;
    if      (stage == 0) { rows = HID_;  K = IN_;  src = ws + WS_Q1; woff = L_W0; }
    else if (stage == 1) { rows = HID_;  K = HID_; src = ws + WS_Q2; woff = L_W_S; }
    else if (stage == 2) { rows = HID_;  K = HID_; src = ws + WS_Q3; woff = L_W_S; }
    else                 { rows = OUTP_; K = HID_; src = ws + WS_Q4; woff = L_W_S; }
    for (int row = tid >> 5; row < rows; row += NWAVES)
      for (int k = (tid & 31) * 8; k < K; k += 32 * 8)
        *(u64*)(smem + woff + row * (K + 16) + k) = ld64(src + row * K + k);
  }
  if (stage == 0)
    for (int c = tid; c < B_ * IN_; c += NTHREADS)
      ((float*)(smem + L_CHM))[c] = 0.f;
  __syncthreads();

  float* outSpk1 = out;
  float* outSpk2 = out + (size_t)T_ * B_ * HID_;
  float* outSpk3 = out + 2 * (size_t)T_ * B_ * HID_;
  float* outSpk4 = out + 3 * (size_t)T_ * B_ * HID_;
  float* outMem4 = outSpk4 + (size_t)T_ * B_ * OUT_;

  float mem[4][8];
#pragma unroll
  for (int i = 0; i < 4; ++i)
#pragma unroll
    for (int j = 0; j < 8; ++j) mem[i][j] = 0.f;

  const int NTICK = T_ + 3;
  for (int tick = 0; tick < NTICK; ++tick) {
    const int t = tick - stage;
    if (t >= 0 && t < T_) {
      const int par = t & 1;
      if (stage == 0) {
        float* chm = (float*)(smem + L_CHM);
        u8*    chs = (u8*)(smem + L_CHSPK);
#pragma unroll 4
        for (int j = 0; j < B_ * IN_ / NTHREADS; ++j) {
          const int c = j * NTHREADS + tid;
          const int b = c >> 8, i = c & 255;
          const float cur = x[b * T_ + t] * Wch[i];
          const float bb  = fminf(fmaxf(cbetas[i], 0.f), 1.f);
          float mv = chm[c];
          const float reset = (mv > 1.f) ? 1.f : 0.f;
          mv = bb * mv + cur - reset;
          chm[c] = mv;
          chs[b * CH_STRIDE + i] = (u8)(mv > 1.f ? 1 : 0);
        }
        if (tid < B_ && t + 1 < T_)
          __builtin_prefetch(&x[tid * T_ + t + 1], 0, 1);
        __syncthreads();
        if (tid < B_) {
          const u64* r = (const u64*)(smem + L_CHSPK + tid * CH_STRIDE);
          unsigned s = 0;
#pragma unroll
          for (int q = 0; q < IN_ / 8; ++q)
            s += (unsigned)((r[q] * 0x0101010101010101ull) >> 56);
          S[tid] = (float)s;
        }
        __syncthreads();
        layer_step<IN_, 32>(smem, L_CHSPK, CH_STRIDE, L_W0, S,
                            b1, 1.0f, mem, t, outSpk1, nullptr, HID_,
                            ws + WS_S1 + par * B_ * HID_);
      } else {
        const u8* inbuf; u8* nb = nullptr; float* os; float* om = nullptr;
        float bet; int Hv;
        if (stage == 1) {
          inbuf = ws + WS_S1 + par * B_ * HID_; nb = ws + WS_S2 + par * B_ * HID_;
          os = outSpk2; bet = b2; Hv = HID_;
        } else if (stage == 2) {
          inbuf = ws + WS_S2 + par * B_ * HID_; nb = ws + WS_S3 + par * B_ * HID_;
          os = outSpk3; bet = b3; Hv = HID_;
        } else {
          inbuf = ws + WS_S3 + par * B_ * HID_;
          os = outSpk4; om = outMem4; bet = b4; Hv = OUT_;
        }
        {
          const int row = tid >> 4;
          const int off = (tid & 15) * 32;
          const u64* src = (const u64*)(inbuf + row * HID_ + off);
          u64* dst = (u64*)(smem + L_ASPK + row * A_STRIDE + off);
          dst[0] = src[0]; dst[1] = src[1]; dst[2] = src[2]; dst[3] = src[3];
        }
        __syncthreads();
        if (tid < B_) {
          const u64* r = (const u64*)(smem + L_ASPK + tid * A_STRIDE);
          unsigned s = 0;
#pragma unroll
          for (int q = 0; q < HID_ / 8; ++q)
            s += (unsigned)((r[q] * 0x0101010101010101ull) >> 56);
          S[tid] = (float)s;
        }
        __syncthreads();
        if (stage == 3)
          layer_step<HID_, 3>(smem, L_ASPK, A_STRIDE, L_W_S, S, bet, 1.0f,
                              mem, t, os, om, Hv, nullptr);
        else
          layer_step<HID_, 32>(smem, L_ASPK, A_STRIDE, L_W_S, S, bet, 1.0f,
                               mem, t, os, nullptr, Hv, nb);
      }
    }

    __builtin_amdgcn_fence(__ATOMIC_RELEASE, "agent");
    __syncthreads();
    if (tid == 0) {
      __hip_atomic_fetch_add(bar, 1u, __ATOMIC_RELAXED, __HIP_MEMORY_SCOPE_AGENT);
      const unsigned target = (unsigned)(tick + 1) * 4u;
      while (__hip_atomic_load(bar, __ATOMIC_RELAXED, __HIP_MEMORY_SCOPE_AGENT) < target)
        __builtin_amdgcn_s_sleep(2);
    }
    __syncthreads();
    __builtin_amdgcn_fence(__ATOMIC_ACQUIRE, "agent");
  }
}

extern "C" void kernel_launch(void* const* d_in, const int* in_sizes, int n_in,
                              void* d_out, int out_size, void* d_ws, size_t ws_size,
                              hipStream_t stream)
{
  const float* x      = (const float*)d_in[0];
  const float* Wch    = (const float*)d_in[1];
  const float* W1     = (const float*)d_in[2];
  const float* W2     = (const float*)d_in[3];
  const float* W3     = (const float*)d_in[4];
  const float* W4     = (const float*)d_in[5];
  const float* cbetas = (const float*)d_in[6];
  const float* pb1    = (const float*)d_in[7];
  const float* pb2    = (const float*)d_in[8];
  const float* pb3    = (const float*)d_in[9];
  const float* pb4    = (const float*)d_in[10];
  u8*    ws  = (u8*)d_ws;
  float* out = (float*)d_out;
  if (ws_size < (size_t)WS_END) return;

  (void)hipFuncSetAttribute((const void*)snn_pipeline,
                            hipFuncAttributeMaxDynamicSharedMemorySize, LDS_BYTES);

  quant_kernel<<<64, 256, 0, stream>>>(W1, W2, W3, W4, ws);
  snn_pipeline<<<4, NTHREADS, LDS_BYTES, stream>>>(x, Wch, cbetas,
                                                   pb1, pb2, pb3, pb4, ws, out);
}
